// MSAAttention_20014547599619
// MI455X (gfx1250) — compile-verified, matrix operand order checked by reading, not run on hardware
//
#include <hip/hip_runtime.h>
#include <hip/hip_bf16.h>

#define DMODEL 512
#define NHEADS 8
#define HDIM   64
#define SEQ    64
#define LEN    512
#define MROWS  (SEQ * LEN)      // 32768
#define ATT_SCALE 0.125f        // 1/sqrt(64)

typedef __bf16 bf16;
typedef __attribute__((ext_vector_type(16))) __bf16 v16bf;
typedef __attribute__((ext_vector_type(8)))  __bf16 v8bf;
typedef __attribute__((ext_vector_type(8)))  float  v8f;

// ---------------------------------------------------------------------------
// Async global -> LDS copies (CDNA5 GLOBAL_LOAD_ASYNC_TO_LDS_B128, ASYNCcnt).
// LDS address = low 32 bits of the generic pointer to a __shared__ object.
// INST_OFFSET applies to BOTH the LDS and global addresses (ISA 15.18.3), so a
// 32B pair shares one LDS-address VGPR and one 64-bit global address.
// ---------------------------------------------------------------------------
__device__ __forceinline__ unsigned lds_addr_of(const void* p) {
  return (unsigned)(unsigned long long)p;
}

__device__ __forceinline__ void async_copy_pair(unsigned lds_addr,
                                                const void* gaddr) {
  asm volatile("global_load_async_to_lds_b128 %0, %1, off\n\t"
               "global_load_async_to_lds_b128 %0, %1, off offset:16"
               :: "v"(lds_addr), "v"(gaddr)
               : "memory");
}

__device__ __forceinline__ void wait_async0() {
  asm volatile("s_wait_asynccnt 0x0" ::: "memory");
}

// ---------------------------------------------------------------------------
// WMMA fragment load from a row-major (row, K) tile in LDS.
// A-matrix 16x32 bf16 layout (ISA 7.12.2): lanes 0-15 hold rows 0-15 with
// K = {0..7, 16..23}; lanes 16-31 hold rows 0-15 with K = {8..15, 24..31}.
// B fragment mirrors this with "row" = output column (contiguous K).
// ---------------------------------------------------------------------------
__device__ __forceinline__ v16bf frag_ld(const bf16* base, int stride) {
  const int lane = threadIdx.x & 31;
  const bf16* p = base + (lane & 15) * stride + ((lane >> 4) << 3);
  v8bf lo = *(const v8bf*)(p);
  v8bf hi = *(const v8bf*)(p + 16);
  v16bf r;
#pragma unroll
  for (int i = 0; i < 8; ++i) { r[i] = lo[i]; r[i + 8] = hi[i]; }
  return r;
}

__device__ __forceinline__ v8f wmma_bf16(v16bf a, v16bf b, v8f c) {
  return __builtin_amdgcn_wmma_f32_16x16x32_bf16(false, a, false, b,
                                                 (short)0, c, false, false);
}

// ---------------------------------------------------------------------------
// fp32 -> bf16 conversion
// ---------------------------------------------------------------------------
__global__ void f32_to_bf16_kernel(const float* __restrict__ in,
                                   bf16* __restrict__ out, int n) {
  int i = blockIdx.x * blockDim.x + threadIdx.x;
  if (i < n) out[i] = (bf16)in[i];
}

// ---------------------------------------------------------------------------
// Y = X @ W^T + b ; X:(M,512) bf16 row-major, W:(512,512) bf16 row-major (N,K)
// Block: 256 threads (8 waves), tile 128(M) x 128(N), K staged 32-wide with
// double-buffered LDS filled by async global->LDS copies. Wave w owns a
// 32(M) x 64(N) sub-tile: 2 A-frags + 4 B-frags -> 8 WMMAs per K-step.
// ---------------------------------------------------------------------------
template <bool OUT_F32>
__global__ __launch_bounds__(256)
void gemm_xwT_kernel(const bf16* __restrict__ X, const bf16* __restrict__ W,
                     const float* __restrict__ bias,
                     bf16* __restrict__ Yb, float* __restrict__ Yf) {
  __shared__ bf16 As[2][128 * 40];   // stride 40 (80B rows, 16B aligned)
  __shared__ bf16 Bs[2][128 * 40];

  const int tid  = threadIdx.x;
  const int wave = tid >> 5, lane = tid & 31;
  const int m0 = blockIdx.x * 128;
  const int n0 = blockIdx.y * 128;
  const int mw = (wave & 3) * 32;    // wave M offset inside block tile
  const int nw = (wave >> 2) * 64;   // wave N offset inside block tile

  v8f z = {};
  v8f acc[2][4];
#pragma unroll
  for (int mt = 0; mt < 2; ++mt)
#pragma unroll
    for (int nt = 0; nt < 4; ++nt) acc[mt][nt] = z;

  // staging: 128 rows x 32 K per tile, 256 threads x 16 bf16 (one 32B pair)
  const int ldr = tid >> 1, ldc = (tid & 1) * 16;
  const bf16* gx = X + (size_t)(m0 + ldr) * DMODEL + ldc;
  const bf16* gw = W + (size_t)(n0 + ldr) * DMODEL + ldc;
  const unsigned a_lds[2] = { lds_addr_of(&As[0][ldr * 40 + ldc]),
                              lds_addr_of(&As[1][ldr * 40 + ldc]) };
  const unsigned b_lds[2] = { lds_addr_of(&Bs[0][ldr * 40 + ldc]),
                              lds_addr_of(&Bs[1][ldr * 40 + ldc]) };

  async_copy_pair(a_lds[0], gx);
  async_copy_pair(b_lds[0], gw);
  wait_async0();
  __syncthreads();

  for (int i = 0; i < 16; ++i) {
    const int cur = i & 1, nxt = cur ^ 1;
    if (i < 15) {   // async-fill next buffer while WMMAs run on current
      const int k = (i + 1) * 32;
      async_copy_pair(a_lds[nxt], gx + k);
      async_copy_pair(b_lds[nxt], gw + k);
    }
    v16bf a0 = frag_ld(&As[cur][(mw)      * 40], 40);
    v16bf a1 = frag_ld(&As[cur][(mw + 16) * 40], 40);
#pragma unroll
    for (int nt = 0; nt < 4; ++nt) {
      v16bf b = frag_ld(&Bs[cur][(nw + nt * 16) * 40], 40);
      acc[0][nt] = wmma_bf16(a0, b, acc[0][nt]);
      acc[1][nt] = wmma_bf16(a1, b, acc[1][nt]);
    }
    wait_async0();
    __syncthreads();
  }

  // epilogue: C/D layout -> VGPR r holds row (r + 8*(lane>=16)), col = lane&15
  const int hi8 = (lane >> 4) * 8, cl = lane & 15;
#pragma unroll
  for (int mt = 0; mt < 2; ++mt) {
#pragma unroll
    for (int nt = 0; nt < 4; ++nt) {
      const int col = n0 + nw + nt * 16 + cl;
      const float bv = bias[col];
#pragma unroll
      for (int r = 0; r < 8; ++r) {
        const int row = m0 + mw + mt * 16 + r + hi8;
        float v = acc[mt][nt][r] + bv;
        if (OUT_F32) Yf[(size_t)row * DMODEL + col] = v;
        else         Yb[(size_t)row * DMODEL + col] = (bf16)v;
      }
    }
  }
}

// ---------------------------------------------------------------------------
// Row attention: block = (l-tile of 64 queries, head h, row s).
// Flash-style online softmax over 8 key chunks of 64.
// ---------------------------------------------------------------------------
__global__ __launch_bounds__(256)
void row_attn_kernel(const bf16* __restrict__ Q, const bf16* __restrict__ Kb,
                     const bf16* __restrict__ Vb,
                     const unsigned char* __restrict__ mask,
                     float* __restrict__ row_out) {
  __shared__ bf16 Qs[64 * 72];
  __shared__ bf16 Ks[64 * 72];
  __shared__ bf16 Vts[64 * 72];     // V transposed: Vts[d][n]
  __shared__ float Ss[64 * 68];
  __shared__ bf16 Ps[64 * 72];
  __shared__ float rmax[64], rsum[64], ralpha[64];

  const int tid = threadIdx.x, wave = tid >> 5, lane = tid & 31;
  const int l0 = blockIdx.x * 64;
  const int h  = blockIdx.y;
  const int s  = blockIdx.z;
  const size_t rowbase = (size_t)s * LEN;

  const int ldrow = tid >> 2, ldc = (tid & 3) * 16;
  const unsigned k_lds = lds_addr_of(&Ks[ldrow * 72 + ldc]);

  { // async-load 64x64 Q tile
    const bf16* g = Q + (rowbase + l0 + ldrow) * DMODEL + h * HDIM + ldc;
    async_copy_pair(lds_addr_of(&Qs[ldrow * 72 + ldc]), g);
  }
  if (tid < 64) { rmax[tid] = -1e30f; rsum[tid] = 0.f; }

  const int mt = wave & 3, nt0 = (wave >> 2) * 2;   // 2 tiles of 16x16 per wave
  v8f z = {};
  v8f o_acc[2] = {z, z};
  const unsigned char* mrow = mask + (size_t)s * LEN;

  for (int j = 0; j < 8; ++j) {
    { // stage K chunk (async, row-major) and V chunk (transposed via VGPRs)
      const bf16* gk = Kb + (rowbase + j * 64 + ldrow) * DMODEL + h * HDIM + ldc;
      async_copy_pair(k_lds, gk);
      const bf16* gv = Vb + (rowbase + j * 64 + ldrow) * DMODEL + h * HDIM + ldc;
      v8bf v0 = *(const v8bf*)gv;
      v8bf v1 = *(const v8bf*)(gv + 8);
#pragma unroll
      for (int i = 0; i < 8; ++i) {
        Vts[(ldc + i) * 72 + ldrow]     = v0[i];
        Vts[(ldc + 8 + i) * 72 + ldrow] = v1[i];
      }
    }
    wait_async0();
    __syncthreads();

    // S = Q @ K^T for this 64x64 chunk
    v8f sacc[2] = {z, z};
#pragma unroll
    for (int ks = 0; ks < 64; ks += 32) {
      v16bf a = frag_ld(&Qs[mt * 16 * 72 + ks], 72);
#pragma unroll
      for (int t = 0; t < 2; ++t) {
        v16bf b = frag_ld(&Ks[(nt0 + t) * 16 * 72 + ks], 72);
        sacc[t] = wmma_bf16(a, b, sacc[t]);
      }
    }
    {
      const int hi8 = (lane >> 4) * 8, cl = lane & 15;
#pragma unroll
      for (int t = 0; t < 2; ++t) {
        const int col = (nt0 + t) * 16 + cl;
        const bool mOK = mrow[j * 64 + col] != 0;
#pragma unroll
        for (int r = 0; r < 8; ++r) {
          const int row = mt * 16 + r + hi8;
          Ss[row * 68 + col] = mOK ? sacc[t][r] * ATT_SCALE : -1e9f;
        }
      }
    }
    __syncthreads();

    // online softmax: 4 lockstep lanes per score row, quad shuffle reductions
    {
      const int row = tid >> 2, seg = tid & 3;
      const float* srow = &Ss[row * 68 + seg * 16];
      float lmax = -1e30f;
#pragma unroll
      for (int c = 0; c < 16; ++c) lmax = fmaxf(lmax, srow[c]);
      lmax = fmaxf(lmax, __shfl_xor(lmax, 1, 32));
      lmax = fmaxf(lmax, __shfl_xor(lmax, 2, 32));
      const float mold = rmax[row];
      const float mn = fmaxf(mold, lmax);
      const float alpha = __expf(mold - mn);
      float sum = 0.f;
      bf16* prow = &Ps[row * 72 + seg * 16];
#pragma unroll
      for (int c = 0; c < 16; ++c) {
        float p = __expf(srow[c] - mn);
        prow[c] = (bf16)p;
        sum += p;
      }
      sum += __shfl_xor(sum, 1, 32);
      sum += __shfl_xor(sum, 2, 32);
      if (seg == 0) {
        rsum[row] = rsum[row] * alpha + sum;
        rmax[row] = mn;
        ralpha[row] = alpha;
      }
    }
    __syncthreads();

    // O = alpha*O + P @ V
    {
      const int hi8 = (lane >> 4) * 8;
#pragma unroll
      for (int t = 0; t < 2; ++t)
#pragma unroll
        for (int r = 0; r < 8; ++r)
          o_acc[t][r] *= ralpha[mt * 16 + r + hi8];
    }
#pragma unroll
    for (int ks = 0; ks < 64; ks += 32) {
      v16bf a = frag_ld(&Ps[mt * 16 * 72 + ks], 72);
#pragma unroll
      for (int t = 0; t < 2; ++t) {
        v16bf b = frag_ld(&Vts[(nt0 + t) * 16 * 72 + ks], 72);
        o_acc[t] = wmma_bf16(a, b, o_acc[t]);
      }
    }
    __syncthreads();
  }

  const int hi8 = (lane >> 4) * 8, cl = lane & 15;
#pragma unroll
  for (int t = 0; t < 2; ++t) {
    const int col = (nt0 + t) * 16 + cl;        // head-dim column
#pragma unroll
    for (int r = 0; r < 8; ++r) {
      const int row = mt * 16 + r + hi8;
      float denom = rsum[row];
      float v = o_acc[t][r] / (denom > 0.f ? denom : 1.f);
      row_out[(rowbase + l0 + row) * DMODEL + h * HDIM + col] = v;
    }
  }
}

// ---------------------------------------------------------------------------
// Column attention: block = (position l, head h). S=64 -> single pass softmax.
// Fuses combined = row_out + col_out (bf16) for the final projection.
// ---------------------------------------------------------------------------
__global__ __launch_bounds__(256)
void col_attn_kernel(const bf16* __restrict__ Q, const bf16* __restrict__ Kb,
                     const bf16* __restrict__ Vb,
                     const unsigned char* __restrict__ mask,
                     const float* __restrict__ row_out,
                     bf16* __restrict__ combined) {
  __shared__ bf16 Qs[64 * 72];
  __shared__ bf16 Ks[64 * 72];
  __shared__ bf16 Vts[64 * 72];
  __shared__ float Ss[64 * 68];
  __shared__ bf16 Ps[64 * 72];
  __shared__ float rsum[64];

  const int tid = threadIdx.x, wave = tid >> 5, lane = tid & 31;
  const int l = blockIdx.x;
  const int h = blockIdx.y;
  const int ldrow = tid >> 2, ldc = (tid & 3) * 16;   // ldrow = sequence index

  {
    const size_t g = ((size_t)ldrow * LEN + l) * DMODEL + h * HDIM + ldc;
    async_copy_pair(lds_addr_of(&Qs[ldrow * 72 + ldc]), Q + g);
    async_copy_pair(lds_addr_of(&Ks[ldrow * 72 + ldc]), Kb + g);
    v8bf v0 = *(const v8bf*)(Vb + g);
    v8bf v1 = *(const v8bf*)(Vb + g + 8);
#pragma unroll
    for (int i = 0; i < 8; ++i) {
      Vts[(ldc + i) * 72 + ldrow]     = v0[i];
      Vts[(ldc + 8 + i) * 72 + ldrow] = v1[i];
    }
  }
  wait_async0();
  __syncthreads();

  const int mt = wave & 3, nt0 = (wave >> 2) * 2;
  v8f z = {};
  v8f sacc[2] = {z, z};
#pragma unroll
  for (int ks = 0; ks < 64; ks += 32) {
    v16bf a = frag_ld(&Qs[mt * 16 * 72 + ks], 72);
#pragma unroll
    for (int t = 0; t < 2; ++t) {
      v16bf b = frag_ld(&Ks[(nt0 + t) * 16 * 72 + ks], 72);
      sacc[t] = wmma_bf16(a, b, sacc[t]);
    }
  }
  const int hi8 = (lane >> 4) * 8, cl = lane & 15;
#pragma unroll
  for (int t = 0; t < 2; ++t) {
    const int col = (nt0 + t) * 16 + cl;                // key sequence index
    const bool mOK = mask[(size_t)col * LEN + l] != 0;  // mask_t[l][col]
#pragma unroll
    for (int r = 0; r < 8; ++r) {
      const int row = mt * 16 + r + hi8;
      Ss[row * 68 + col] = mOK ? sacc[t][r] * ATT_SCALE : -1e9f;
    }
  }
  __syncthreads();

  { // softmax: 4 lockstep lanes per row
    const int row = tid >> 2, seg = tid & 3;
    const float* srow = &Ss[row * 68 + seg * 16];
    float lmax = -1e30f;
#pragma unroll
    for (int c = 0; c < 16; ++c) lmax = fmaxf(lmax, srow[c]);
    lmax = fmaxf(lmax, __shfl_xor(lmax, 1, 32));
    lmax = fmaxf(lmax, __shfl_xor(lmax, 2, 32));
    float sum = 0.f;
    bf16* prow = &Ps[row * 72 + seg * 16];
#pragma unroll
    for (int c = 0; c < 16; ++c) {
      float p = __expf(srow[c] - lmax);
      prow[c] = (bf16)p;
      sum += p;
    }
    sum += __shfl_xor(sum, 1, 32);
    sum += __shfl_xor(sum, 2, 32);
    if (seg == 0) rsum[row] = sum;
  }
  __syncthreads();

  v8f oacc[2] = {z, z};
#pragma unroll
  for (int ks = 0; ks < 64; ks += 32) {
    v16bf a = frag_ld(&Ps[mt * 16 * 72 + ks], 72);
#pragma unroll
    for (int t = 0; t < 2; ++t) {
      v16bf b = frag_ld(&Vts[(nt0 + t) * 16 * 72 + ks], 72);
      oacc[t] = wmma_bf16(a, b, oacc[t]);
    }
  }
#pragma unroll
  for (int t = 0; t < 2; ++t) {
    const int col = (nt0 + t) * 16 + cl;   // head dim
#pragma unroll
    for (int r = 0; r < 8; ++r) {
      const int srow = mt * 16 + r + hi8;  // sequence index
      float denom = rsum[srow];
      float v = oacc[t][r] / (denom > 0.f ? denom : 1.f);
      size_t idx = ((size_t)srow * LEN + l) * DMODEL + h * HDIM + col;
      combined[idx] = (bf16)(row_out[idx] + v);
    }
  }
}

// ---------------------------------------------------------------------------
// Host orchestration
// ---------------------------------------------------------------------------
extern "C" void kernel_launch(void* const* d_in, const int* in_sizes, int n_in,
                              void* d_out, int out_size, void* d_ws, size_t ws_size,
                              hipStream_t stream) {
  (void)in_sizes; (void)n_in; (void)out_size; (void)ws_size;

  const size_t act_elems = (size_t)MROWS * DMODEL;      // 16,777,216
  const size_t w_elems   = (size_t)DMODEL * DMODEL;     // 262,144

  char* ws = (char*)d_ws;
  size_t off = 0;
  bf16* Xb = (bf16*)(ws + off);        off += act_elems * sizeof(bf16);
  bf16* Wb = (bf16*)(ws + off);        off += 7 * w_elems * sizeof(bf16);
  bf16* proj[6];
  for (int i = 0; i < 6; ++i) { proj[i] = (bf16*)(ws + off); off += act_elems * sizeof(bf16); }
  float* row_out = (float*)(ws + off); off += act_elems * sizeof(float);
  bf16* combined = (bf16*)(ws + off);  off += act_elems * sizeof(bf16);

  // 1) convert activations + weights to bf16
  {
    int n = (int)act_elems;
    f32_to_bf16_kernel<<<(n + 255) / 256, 256, 0, stream>>>(
        (const float*)d_in[0], Xb, n);
  }
  const int widx[7] = {2, 4, 6, 8, 10, 12, 14};   // rq,rk,rv,cq,ck,cv,out weights
  for (int i = 0; i < 7; ++i) {
    int n = (int)w_elems;
    f32_to_bf16_kernel<<<(n + 255) / 256, 256, 0, stream>>>(
        (const float*)d_in[widx[i]], Wb + (size_t)i * w_elems, n);
  }

  // 2) six QKV projections (row q,k,v then col q,k,v)
  const int bidx[6] = {3, 5, 7, 9, 11, 13};
  dim3 ggrid(MROWS / 128, DMODEL / 128);
  for (int i = 0; i < 6; ++i) {
    gemm_xwT_kernel<false><<<ggrid, 256, 0, stream>>>(
        Xb, Wb + (size_t)i * w_elems, (const float*)d_in[bidx[i]],
        proj[i], nullptr);
  }

  const unsigned char* mask = (const unsigned char*)d_in[1];

  // 3) row attention -> row_out (fp32)
  row_attn_kernel<<<dim3(LEN / 64, NHEADS, SEQ), 256, 0, stream>>>(
      proj[0], proj[1], proj[2], mask, row_out);

  // 4) column attention, fused combine -> combined (bf16)
  col_attn_kernel<<<dim3(LEN, NHEADS), 256, 0, stream>>>(
      proj[3], proj[4], proj[5], mask, row_out, combined);

  // 5) output projection -> d_out (fp32)
  gemm_xwT_kernel<true><<<ggrid, 256, 0, stream>>>(
      combined, Wb + 6 * w_elems, (const float*)d_in[15],
      nullptr, (float*)d_out);
}
